// DocLSTM_7078106104352
// MI455X (gfx1250) — matrix operand order repaired and checked by reading, not compiled, not run on hardware
//
#include <hip/hip_runtime.h>
#include <stdint.h>
#include <stddef.h>

// ---------------------------------------------------------------------------
// Types for CDNA5 WMMA (gfx1250, wave32)
// ---------------------------------------------------------------------------
typedef __bf16 bf16;
typedef __attribute__((ext_vector_type(16))) __bf16 v16bf;
typedef __attribute__((ext_vector_type(8)))  __bf16 v8bf;
typedef __attribute__((ext_vector_type(8)))  float  v8f;

#define MDIM 1024      // mem_dim
#define DDIM 512       // in_dim
#define BS   144       // sentence batch padded (1 + 8*16 = 129 -> 144)
#define BROWS_S 129
#define BP   16        // para/body batch padded
#define WLEN 64
#define NSENT 16
#define NPARA 8

// ---------------------------------------------------------------------------
// WMMA fragment loads (layouts per CDNA5 ISA 7.12.2, bf16 16x32 A / 32x16 B)
// A: lanes 0-15 hold rows M=0..15 with K = {k0..k0+7, k0+16..k0+23}, k0=0;
//    lanes 16-31 same rows but k0=8.  Two contiguous 8-element chunks.
// B: lane n%16 holds column N=n with K = {k0..k0+7, k0+16..k0+23},
//    k0=0 for lanes 0-15, k0=8 for lanes 16-31 (same as A).
// Both reduce to two 16-byte contiguous loads from row-major storage.
// ---------------------------------------------------------------------------
__device__ __forceinline__ v16bf load_frag_A(const bf16* __restrict__ base, int ld,
                                             int mbase, int kbase, int lane) {
  const int row = mbase + (lane & 15);
  const int k0  = kbase + ((lane & 16) ? 8 : 0);
  const bf16* p = base + (size_t)row * ld + k0;
  union { v16bf v; v8bf h[2]; } u;
  u.h[0] = *(const v8bf*)(p);        // K = k0 .. k0+7    -> VGPRs 0..3
  u.h[1] = *(const v8bf*)(p + 16);   // K = k0+16..k0+23  -> VGPRs 4..7
  return u.v;
}

__device__ __forceinline__ v16bf load_frag_B(const bf16* __restrict__ W, int ld,
                                             int nbase, int kbase, int lane) {
  const int n  = nbase + (lane & 15);
  const int k0 = kbase + ((lane & 16) ? 8 : 0);
  const bf16* p = W + (size_t)n * ld + k0;   // two 8-element chunks
  union { v16bf v; v8bf h[2]; } u;
  u.h[0] = *(const v8bf*)(p);
  u.h[1] = *(const v8bf*)(p + 16);
  return u.v;
}

// ---------------------------------------------------------------------------
// Fused LSTM step:  gates = X_t @ Wih^T + h_in @ Whh^T + bias ; cell update.
// Block = 128 threads = 4 waves; wave g computes gate g (i,f,g,o order) for
// the same 16x16 (batch x hidden) tile.  Gate tiles meet in LDS for the
// elementwise cell update.
// grid.x = MDIM/16 hidden tiles, grid.y = Bpad/16 batch tiles.
// ---------------------------------------------------------------------------
__global__ __launch_bounds__(128)
void lstm_step_kernel(const bf16* __restrict__ Xt, int Din,
                      const bf16* __restrict__ Wih,    // [4*MDIM][Din]
                      const bf16* __restrict__ Whh,    // [4*MDIM][MDIM]
                      const bf16* __restrict__ h_in,   // [Bpad][MDIM]
                      const float* __restrict__ bias,  // [4*MDIM]
                      float* __restrict__ c,           // [Bpad][MDIM] (in-place)
                      bf16*  __restrict__ h_out_bf,    // [Bpad][MDIM]
                      float* __restrict__ h_out_f32,   // [Bpad][MDIM]
                      int Brows)
{
  const int lane  = threadIdx.x & 31;
  const int wave  = threadIdx.x >> 5;          // gate index 0..3
  const int nblk  = blockIdx.x;                // hidden tile
  const int mbase = blockIdx.y << 4;           // batch row base
  const int ngate = wave * MDIM + (nblk << 4); // row base inside W

  v8f acc = {};

  // x_t contribution
  for (int k = 0; k < Din; k += 32) {
    if (k + 64 < Din)
      __builtin_prefetch(Wih + (size_t)ngate * Din + k + 64, 0, 1);
    v16bf a = load_frag_A(Xt,  Din, mbase, k, lane);
    v16bf b = load_frag_B(Wih, Din, ngate, k, lane);
    acc = __builtin_amdgcn_wmma_f32_16x16x32_bf16(false, a, false, b,
                                                  (short)0, acc, false, false);
  }
  // recurrent contribution
  for (int k = 0; k < MDIM; k += 32) {
    if (k + 64 < MDIM)
      __builtin_prefetch(Whh + (size_t)ngate * MDIM + k + 64, 0, 1);
    v16bf a = load_frag_A(h_in, MDIM, mbase, k, lane);
    v16bf b = load_frag_B(Whh,  MDIM, ngate, k, lane);
    acc = __builtin_amdgcn_wmma_f32_16x16x32_bf16(false, a, false, b,
                                                  (short)0, acc, false, false);
  }

  // C/D layout: lane l, VGPR v -> (M = v + (l>=16 ? 8 : 0), N = l & 15)
  __shared__ float g_lds[4][16][16];
  {
    const int mrow = (lane >> 4) << 3;
    const int ncol = lane & 15;
#pragma unroll
    for (int v = 0; v < 8; ++v) g_lds[wave][mrow + v][ncol] = acc[v];
  }
  __syncthreads();

  // elementwise LSTM cell update over the 16x16 tile (2 elems per thread)
  for (int e = threadIdx.x; e < 256; e += 128) {
    const int m = e >> 4, n = e & 15;
    const int row = mbase + m;
    if (row >= Brows) continue;
    const int col = (nblk << 4) + n;
    const float gi = g_lds[0][m][n] + bias[0 * MDIM + col];
    const float gf = g_lds[1][m][n] + bias[1 * MDIM + col];
    const float gg = g_lds[2][m][n] + bias[2 * MDIM + col];
    const float go = g_lds[3][m][n] + bias[3 * MDIM + col];
    const float si = 1.0f / (1.0f + expf(-gi));
    const float sf = 1.0f / (1.0f + expf(-gf));
    const float so = 1.0f / (1.0f + expf(-go));
    const float tg = tanhf(gg);
    const size_t idx = (size_t)row * MDIM + col;
    const float cn = sf * c[idx] + si * tg;
    c[idx] = cn;
    const float hn = so * tanhf(cn);
    h_out_f32[idx] = hn;
    h_out_bf[idx]  = (bf16)hn;
  }
}

// ---------------------------------------------------------------------------
// Prep / glue kernels
// ---------------------------------------------------------------------------
__global__ void zero_u32_kernel(uint32_t* p, size_t n) {
  size_t i = (size_t)blockIdx.x * blockDim.x + threadIdx.x;
  if (i < n) p[i] = 0u;
}

__global__ void cvt_f32_bf16_kernel(const float* __restrict__ s,
                                    bf16* __restrict__ d, int n) {
  int i = blockIdx.x * blockDim.x + threadIdx.x;
  if (i < n) d[i] = (bf16)s[i];
}

__global__ void bias_combine_kernel(const float* __restrict__ a,
                                    const float* __restrict__ b,
                                    float* __restrict__ d, int n) {
  int i = blockIdx.x * blockDim.x + threadIdx.x;
  if (i < n) d[i] = a[i] + b[i];
}

// Xs[t][b][d] = bf16(emb[tok(b,t)][d]) ; pad rows (b>=129) = 0
__global__ void embed_gather_kernel(const int* __restrict__ headline,
                                    const int* __restrict__ body,   // [P*S*W]
                                    const float* __restrict__ emb,
                                    bf16* __restrict__ Xs) {
  const size_t total = (size_t)WLEN * BS * DDIM;
  size_t i = (size_t)blockIdx.x * blockDim.x + threadIdx.x;
  if (i >= total) return;
  const int d = (int)(i & (DDIM - 1));
  const int b = (int)((i >> 9) % BS);
  const int t = (int)(i / ((size_t)DDIM * BS));
  float val = 0.0f;
  if (b < BROWS_S) {
    const int tok = (b == 0) ? headline[t] : body[(b - 1) * WLEN + t];
    val = emb[(size_t)tok * DDIM + d];
  }
  Xs[i] = (bf16)val;
}

// Xp[s][p][d] = sentence final h of (para p, sent s); b = 1 + p*16 + s
__global__ void repack_para_kernel(const bf16* __restrict__ hs,
                                   bf16* __restrict__ Xp) {
  const size_t total = (size_t)NSENT * BP * MDIM;
  size_t i = (size_t)blockIdx.x * blockDim.x + threadIdx.x;
  if (i >= total) return;
  const int d = (int)(i & (MDIM - 1));
  const int p = (int)((i >> 10) & 15);
  const int s = (int)(i >> 14);
  Xp[i] = (p < NPARA) ? hs[(size_t)(1 + p * NSENT + s) * MDIM + d] : (bf16)0.0f;
}

// Xb[t][0][d] = para final h of paragraph t
__global__ void repack_body_kernel(const bf16* __restrict__ hp,
                                   bf16* __restrict__ Xb) {
  const size_t total = (size_t)NPARA * BP * MDIM;
  size_t i = (size_t)blockIdx.x * blockDim.x + threadIdx.x;
  if (i >= total) return;
  const int d = (int)(i & (MDIM - 1));
  const int r = (int)((i >> 10) & 15);
  const int t = (int)(i >> 14);
  Xb[i] = (r == 0) ? hp[(size_t)t * MDIM + d] : (bf16)0.0f;
}

// out[0:1024] = h_body (row 0), out[1024:2048] = r_hidden (sentence row 0)
__global__ void write_out_kernel(const float* __restrict__ hb,
                                 const float* __restrict__ hs,
                                 float* __restrict__ out) {
  int i = blockIdx.x * blockDim.x + threadIdx.x;
  if (i < MDIM) out[i] = hb[i];
  else if (i < 2 * MDIM) out[i] = hs[i - MDIM];
}

// ---------------------------------------------------------------------------
// Host-side orchestration
// ---------------------------------------------------------------------------
extern "C" void kernel_launch(void* const* d_in, const int* in_sizes, int n_in,
                              void* d_out, int out_size, void* d_ws, size_t ws_size,
                              hipStream_t stream) {
  (void)in_sizes; (void)n_in; (void)out_size; (void)ws_size;

  const int*   headline = (const int*)  d_in[0];
  const int*   body_tok = (const int*)  d_in[1];
  const float* emb      = (const float*)d_in[2];
  const float* sW_ih = (const float*)d_in[3];
  const float* sW_hh = (const float*)d_in[4];
  const float* sb_ih = (const float*)d_in[5];
  const float* sb_hh = (const float*)d_in[6];
  const float* pW_ih = (const float*)d_in[7];
  const float* pW_hh = (const float*)d_in[8];
  const float* pb_ih = (const float*)d_in[9];
  const float* pb_hh = (const float*)d_in[10];
  const float* bW_ih = (const float*)d_in[11];
  const float* bW_hh = (const float*)d_in[12];
  const float* bb_ih = (const float*)d_in[13];
  const float* bb_hh = (const float*)d_in[14];
  float* out = (float*)d_out;

  uint8_t* ws = (uint8_t*)d_ws;
  size_t off = 0;
  auto alloc = [&](size_t bytes) -> uint8_t* {
    uint8_t* p = ws + off;
    off = (off + bytes + 255) & ~(size_t)255;
    return p;
  };

  // ---- zero-initialized block (h0, c0 state) ----
  const size_t zstart = off;
  bf16*  hs_bf = (bf16*) alloc((size_t)2 * BS * MDIM * sizeof(bf16)); // double buf
  bf16*  hp_bf = (bf16*) alloc((size_t)2 * BP * MDIM * sizeof(bf16));
  bf16*  hb_bf = (bf16*) alloc((size_t)2 * BP * MDIM * sizeof(bf16));
  float* cs    = (float*)alloc((size_t)BS * MDIM * sizeof(float));
  float* cp    = (float*)alloc((size_t)BP * MDIM * sizeof(float));
  float* cb    = (float*)alloc((size_t)BP * MDIM * sizeof(float));
  const size_t zend = off;

  // ---- non-zeroed scratch ----
  float* hs_f32 = (float*)alloc((size_t)2 * BS * MDIM * sizeof(float));
  float* hp_f32 = (float*)alloc((size_t)2 * BP * MDIM * sizeof(float));
  float* hb_f32 = (float*)alloc((size_t)2 * BP * MDIM * sizeof(float));
  bf16* sWih_bf = (bf16*)alloc((size_t)4 * MDIM * DDIM * sizeof(bf16));
  bf16* sWhh_bf = (bf16*)alloc((size_t)4 * MDIM * MDIM * sizeof(bf16));
  bf16* pWih_bf = (bf16*)alloc((size_t)4 * MDIM * MDIM * sizeof(bf16));
  bf16* pWhh_bf = (bf16*)alloc((size_t)4 * MDIM * MDIM * sizeof(bf16));
  bf16* bWih_bf = (bf16*)alloc((size_t)4 * MDIM * MDIM * sizeof(bf16));
  bf16* bWhh_bf = (bf16*)alloc((size_t)4 * MDIM * MDIM * sizeof(bf16));
  float* sbias  = (float*)alloc((size_t)4 * MDIM * sizeof(float));
  float* pbias  = (float*)alloc((size_t)4 * MDIM * sizeof(float));
  float* bbias  = (float*)alloc((size_t)4 * MDIM * sizeof(float));
  bf16* Xs = (bf16*)alloc((size_t)WLEN  * BS * DDIM * sizeof(bf16));
  bf16* Xp = (bf16*)alloc((size_t)NSENT * BP * MDIM * sizeof(bf16));
  bf16* Xb = (bf16*)alloc((size_t)NPARA * BP * MDIM * sizeof(bf16));

  const int TPB = 256;
  auto blks = [&](size_t n) { return (unsigned)((n + TPB - 1) / TPB); };

  // 1) zero h0/c0 state
  {
    size_t zwords = (zend - zstart) / 4;
    zero_u32_kernel<<<blks(zwords), TPB, 0, stream>>>((uint32_t*)(ws + zstart), zwords);
  }

  // 2) weight conversion fp32 -> bf16; bias combine
  {
    int n1 = 4 * MDIM * DDIM, n2 = 4 * MDIM * MDIM, nb = 4 * MDIM;
    cvt_f32_bf16_kernel<<<blks(n1), TPB, 0, stream>>>(sW_ih, sWih_bf, n1);
    cvt_f32_bf16_kernel<<<blks(n2), TPB, 0, stream>>>(sW_hh, sWhh_bf, n2);
    cvt_f32_bf16_kernel<<<blks(n2), TPB, 0, stream>>>(pW_ih, pWih_bf, n2);
    cvt_f32_bf16_kernel<<<blks(n2), TPB, 0, stream>>>(pW_hh, pWhh_bf, n2);
    cvt_f32_bf16_kernel<<<blks(n2), TPB, 0, stream>>>(bW_ih, bWih_bf, n2);
    cvt_f32_bf16_kernel<<<blks(n2), TPB, 0, stream>>>(bW_hh, bWhh_bf, n2);
    bias_combine_kernel<<<blks(nb), TPB, 0, stream>>>(sb_ih, sb_hh, sbias, nb);
    bias_combine_kernel<<<blks(nb), TPB, 0, stream>>>(pb_ih, pb_hh, pbias, nb);
    bias_combine_kernel<<<blks(nb), TPB, 0, stream>>>(bb_ih, bb_hh, bbias, nb);
  }

  // 3) embedding gather for sentence LSTM inputs
  {
    size_t n = (size_t)WLEN * BS * DDIM;
    embed_gather_kernel<<<blks(n), TPB, 0, stream>>>(headline, body_tok, emb, Xs);
  }

  // 4) sentence LSTM: 64 steps, B=129 (padded 144), Din=512
  {
    dim3 grid(MDIM / 16, BS / 16);
    for (int t = 0; t < WLEN; ++t) {
      const bf16* xin   = Xs + (size_t)t * BS * DDIM;
      const bf16* hin   = hs_bf  + (size_t)(t & 1) * BS * MDIM;
      bf16*       houtb = hs_bf  + (size_t)((t + 1) & 1) * BS * MDIM;
      float*      houtf = hs_f32 + (size_t)((t + 1) & 1) * BS * MDIM;
      lstm_step_kernel<<<grid, 128, 0, stream>>>(xin, DDIM, sWih_bf, sWhh_bf,
                                                 hin, sbias, cs, houtb, houtf,
                                                 BROWS_S);
    }
  }
  // final sentence state in buffer 0 (64 steps, even)

  // 5) repack sentence outputs -> para LSTM inputs
  {
    size_t n = (size_t)NSENT * BP * MDIM;
    repack_para_kernel<<<blks(n), TPB, 0, stream>>>(hs_bf, Xp);
  }

  // 6) paragraph LSTM: 16 steps, B=8 (padded 16), Din=1024
  {
    dim3 grid(MDIM / 16, BP / 16);
    for (int s = 0; s < NSENT; ++s) {
      const bf16* xin   = Xp + (size_t)s * BP * MDIM;
      const bf16* hin   = hp_bf  + (size_t)(s & 1) * BP * MDIM;
      bf16*       houtb = hp_bf  + (size_t)((s + 1) & 1) * BP * MDIM;
      float*      houtf = hp_f32 + (size_t)((s + 1) & 1) * BP * MDIM;
      lstm_step_kernel<<<grid, 128, 0, stream>>>(xin, MDIM, pWih_bf, pWhh_bf,
                                                 hin, pbias, cp, houtb, houtf,
                                                 NPARA);
    }
  }
  // final para state in buffer 0 (16 steps, even)

  // 7) repack para outputs -> body LSTM inputs
  {
    size_t n = (size_t)NPARA * BP * MDIM;
    repack_body_kernel<<<blks(n), TPB, 0, stream>>>(hp_bf, Xb);
  }

  // 8) body LSTM: 8 steps, B=1 (padded 16), Din=1024
  {
    dim3 grid(MDIM / 16, BP / 16);
    for (int t = 0; t < NPARA; ++t) {
      const bf16* xin   = Xb + (size_t)t * BP * MDIM;
      const bf16* hin   = hb_bf  + (size_t)(t & 1) * BP * MDIM;
      bf16*       houtb = hb_bf  + (size_t)((t + 1) & 1) * BP * MDIM;
      float*      houtf = hb_f32 + (size_t)((t + 1) & 1) * BP * MDIM;
      lstm_step_kernel<<<grid, 128, 0, stream>>>(xin, MDIM, bWih_bf, bWhh_bf,
                                                 hin, bbias, cb, houtb, houtf,
                                                 1);
    }
  }
  // final body state in buffer 0 (8 steps, even)

  // 9) write [h_body ; r_hidden]
  write_out_kernel<<<blks(2 * MDIM), TPB, 0, stream>>>(hb_f32, hs_f32, out);
}
